// DFC_KL_2D_17523466567754
// MI455X (gfx1250) — matrix operand order repaired and checked by reading, not compiled, not run on hardware
//
#include <hip/hip_runtime.h>
#include <hip/hip_bf16.h>

// DFC-KL clustering for MI455X (gfx1250, wave32, WMMA).
// b=8, c=512, n=16384, k=256, 10 stages. The two per-stage GEMMs
// (dis = ent - proto.log_x^T, proto = bm.x) run on v_wmma_f32_16x16x32_bf16,
// with double-buffered GLOBAL_LOAD_ASYNC_TO_LDS_B128 staging (ASYNCcnt).

#define BB 8
#define CC 512
#define NN 16384
#define KK 256
#define STAGES 10

typedef unsigned short ushort_t;
typedef unsigned int uint_t;
typedef __attribute__((ext_vector_type(16))) __bf16 v16bf;
typedef __attribute__((ext_vector_type(8)))  float  v8f;
typedef int gvec4i __attribute__((vector_size(16)));   // matches builtin prototype

// ---- async global->LDS staging (CDNA5). 16B per call, per-lane addresses. ----
#if defined(__HIP_DEVICE_COMPILE__) && __has_builtin(__builtin_amdgcn_global_load_async_to_lds_b128)
#define HAS_ASYNC_LDS 1
#define ASYNC_CP16(g, l)                                                               \
    __builtin_amdgcn_global_load_async_to_lds_b128(                                    \
        (__attribute__((address_space(1))) gvec4i*)(unsigned long long)(g),            \
        (__attribute__((address_space(3))) gvec4i*)(unsigned int)(unsigned long long)(l), \
        0, 0)
#if __has_builtin(__builtin_amdgcn_s_wait_asynccnt)
#define ASYNC_FENCE() __builtin_amdgcn_s_wait_asynccnt(0)
#else
#define ASYNC_FENCE() asm volatile("s_wait_asynccnt 0x0" ::: "memory")
#endif
#else
#define HAS_ASYNC_LDS 0
#define ASYNC_CP16(g, l) (*(uint4*)(l) = *(const uint4*)(g))
#define ASYNC_FENCE() ((void)0)
#endif

__device__ __forceinline__ ushort_t f2bf(float f) {
    uint_t u = __float_as_uint(f);
    u = u + 0x7FFFu + ((u >> 16) & 1u);   // round-to-nearest-even
    return (ushort_t)(u >> 16);
}
__device__ __forceinline__ float bf2f(ushort_t h) {
    return __uint_as_float(((uint_t)h) << 16);
}

// ---------------------------------------------------------------- reductions
__global__ void k_minmax_part(const float* __restrict__ x,
                              float* __restrict__ pmin, float* __restrict__ pmax) {
    int b = blockIdx.y, t = threadIdx.x, blk = blockIdx.x;
    size_t base = (size_t)b * CC * NN;
    size_t total = (size_t)CC * NN;
    float mn = 3.4e38f, mx = -3.4e38f;
    for (size_t i = (size_t)blk * blockDim.x + t; i < total;
         i += (size_t)gridDim.x * blockDim.x) {
        float v = x[base + i];
        mn = fminf(mn, v); mx = fmaxf(mx, v);
    }
    __shared__ float smn[256], smx[256];
    smn[t] = mn; smx[t] = mx; __syncthreads();
    for (int s = 128; s > 0; s >>= 1) {
        if (t < s) { smn[t] = fminf(smn[t], smn[t + s]); smx[t] = fmaxf(smx[t], smx[t + s]); }
        __syncthreads();
    }
    if (t == 0) { pmin[b * 256 + blk] = smn[0]; pmax[b * 256 + blk] = smx[0]; }
}

__global__ void k_minmax_fin(const float* __restrict__ pmin, const float* __restrict__ pmax,
                             float* __restrict__ mnv, float* __restrict__ invv) {
    int b = blockIdx.x, t = threadIdx.x;
    __shared__ float smn[256], smx[256];
    smn[t] = pmin[b * 256 + t]; smx[t] = pmax[b * 256 + t]; __syncthreads();
    for (int s = 128; s > 0; s >>= 1) {
        if (t < s) { smn[t] = fminf(smn[t], smn[t + s]); smx[t] = fmaxf(smx[t], smx[t + s]); }
        __syncthreads();
    }
    if (t == 0) { mnv[b] = smn[0]; invv[b] = 1.0f / (smx[0] - smn[0]); }
}

// normalize, transpose to (b,n,c), make bf16 x / log(x), seed prototypes
__global__ void k_transform(const float* __restrict__ x,
                            const float* __restrict__ mnv, const float* __restrict__ invv,
                            ushort_t* __restrict__ xbf, ushort_t* __restrict__ lxbf,
                            float* __restrict__ protoF, ushort_t* __restrict__ protoH) {
    int b = blockIdx.y;
    size_t idx = (size_t)blockIdx.x * blockDim.x + threadIdx.x;
    if (idx >= (size_t)CC * NN) return;
    int c = (int)(idx >> 14);
    int nn = (int)(idx & (NN - 1));
    float y = (x[(size_t)b * CC * NN + idx] - mnv[b]) * invv[b] + 1e-6f;
    ushort_t yb = f2bf(y);
    size_t tp = ((size_t)b * NN + nn) * CC + c;
    xbf[tp] = yb;
    lxbf[tp] = f2bf(__logf(y));
    if ((nn & 63) == 0) {                     // grid stride n/k = 64
        int kk = nn >> 6;
        size_t pi = ((size_t)b * KK + kk) * CC + c;
        protoF[pi] = y; protoH[pi] = yb;
    }
}

__global__ void k_zero_u16(ushort_t* p, size_t n) {
    size_t i = (size_t)blockIdx.x * blockDim.x + threadIdx.x;
    if (i < n) p[i] = 0;
}
__global__ void k_zero_f32(float* p, size_t n) {
    size_t i = (size_t)blockIdx.x * blockDim.x + threadIdx.x;
    if (i < n) p[i] = 0.0f;
}

// ent[b,k] = sum_c proto * log(proto)  (used once before the stage loop)
__global__ void k_ent(const float* __restrict__ protoF, float* __restrict__ ent) {
    int b = blockIdx.y, kk = blockIdx.x, t = threadIdx.x;
    const float* p = protoF + ((size_t)b * KK + kk) * CC;
    float s = 0.f;
    for (int c = t; c < CC; c += 256) { float v = p[c]; s += v * __logf(v); }
    __shared__ float sh[256];
    sh[t] = s; __syncthreads();
    for (int st = 128; st > 0; st >>= 1) { if (t < st) sh[t] += sh[t + st]; __syncthreads(); }
    if (t == 0) ent[b * KK + kk] = sh[0];
}

// ------------------------------------------------ distance GEMM + argmin + bm
// block: 512 threads (16 waves). Block tile: 256 k x 128 n, K-loop over c with
// double-buffered async global->LDS staging. LDS tiles are fragment-packed per
// the 16-bit WMMA VGPR layouts so each lane reads its fragment as 32 contiguous
// bytes (2x ds_load_b128).
__global__ __launch_bounds__(512)
void k_dist(const ushort_t* __restrict__ protoH, const ushort_t* __restrict__ lxbf,
            const float* __restrict__ ent, int* __restrict__ cluster,
            ushort_t* __restrict__ bmm) {
    int b = blockIdx.y;
    int nbase = blockIdx.x * 128;
    int t = threadIdx.x, lane = t & 31, wave = t >> 5;

    __shared__ __align__(32) ushort_t lA[2][16][32][16];  // 16 k-subtiles (16x32 bf16)
    __shared__ __align__(32) ushort_t lB[2][8][32][16];   // 8 n-subtiles (32x16 bf16)
    __shared__ float lEnt[KK];
    __shared__ float lMin[16][128];
    __shared__ int   lArg[16][128];

    if (t < KK) lEnt[t] = ent[b * KK + t];

    v8f acc[8] = {};
    const size_t pB = (size_t)b * KK * CC;
    const size_t xB = ((size_t)b * NN + nbase) * CC;

    auto stageA = [&](int cc, int buf) {
#pragma unroll
        for (int i = 0; i < 2; i++) {
            int idx = t + i * 512;
            int kk = idx >> 2, cg = idx & 3;
            const ushort_t* src = protoH + pB + (size_t)kk * CC + cc + cg * 8;
            ushort_t* dst = &lA[buf][kk >> 4][(kk & 15) + ((cg & 1) << 4)][(cg >> 1) << 3];
            ASYNC_CP16(src, dst);
        }
    };
    auto stageB = [&](int cc, int buf) {
        int nl = t >> 2, cg = t & 3;
        const ushort_t* src = lxbf + xB + (size_t)nl * CC + cc + cg * 8;
        ushort_t* dst = &lB[buf][nl >> 4][((cg & 1) << 4) + (nl & 15)][(cg >> 1) << 3];
        ASYNC_CP16(src, dst);
    };

    stageA(0, 0); stageB(0, 0);
    ASYNC_FENCE();
    __syncthreads();

    for (int cc = 0; cc < CC; cc += 32) {
        int buf = (cc >> 5) & 1;
        if (cc + 32 < CC) {                 // async-stage next chunk into other buffer
            stageA(cc + 32, buf ^ 1);
            stageB(cc + 32, buf ^ 1);
            __builtin_prefetch(protoH + pB + (size_t)(t >> 1) * CC + cc + 64, 0, 3);
        }
        v16bf a = *(const v16bf*)&lA[buf][wave][lane][0];
#pragma unroll
        for (int ns = 0; ns < 8; ns++) {
            v16bf bb = *(const v16bf*)&lB[buf][ns][lane][0];
            acc[ns] = __builtin_amdgcn_wmma_f32_16x16x32_bf16(
                false, a, false, bb, (short)0, acc[ns], false, false);
        }
        ASYNC_FENCE();
        __syncthreads();
    }

    // dis = ent[k] - acc ; per-column argmin over this wave's 16 k rows
    int kbase = wave * 16;
    int moff = (lane >> 4) * 8;          // C/D layout: lanes>=16 hold M=8..15
    int ncol = lane & 15;
#pragma unroll
    for (int ns = 0; ns < 8; ns++) {
        float best = 3.4e38f; int barg = 0;
#pragma unroll
        for (int r = 0; r < 8; r++) {
            int kk = kbase + moff + r;
            float d = lEnt[kk] - acc[ns][r];
            if (d < best) { best = d; barg = kk; }
        }
        float ob = __shfl_xor(best, 16, 32);
        int   oa = __shfl_xor(barg, 16, 32);
        if (ob < best || (ob == best && oa < barg)) { best = ob; barg = oa; }
        if (lane < 16) { lMin[wave][ns * 16 + ncol] = best; lArg[wave][ns * 16 + ncol] = barg; }
    }
    __syncthreads();
    if (t < 128) {
        float best = lMin[0][t]; int barg = lArg[0][t];
#pragma unroll
        for (int w = 1; w < 16; w++) {
            float v = lMin[w][t]; int a2 = lArg[w][t];
            if (v < best || (v == best && a2 < barg)) { best = v; barg = a2; }
        }
        cluster[(size_t)b * NN + nbase + t] = barg;
        bmm[((size_t)b * KK + barg) * NN + nbase + t] = 0x3F80; // bf16 1.0 (OR-accumulates)
    }
}

// ------------------------------------------- proto update GEMM: acc += bm . x
// block tile: 256 k x 128 c, K-loop over one n-segment; f32 atomics combine
// the 16 n-segments. A staged async, B staged via coalesced load + LDS scatter.
__global__ __launch_bounds__(512)
void k_update(const ushort_t* __restrict__ bmm, const ushort_t* __restrict__ xbf,
              float* __restrict__ protoAcc) {
    int b = blockIdx.z;
    int cbase = blockIdx.x * 128;
    int n0 = blockIdx.y * (NN / 16);
    int t = threadIdx.x, lane = t & 31, wave = t >> 5;

    __shared__ __align__(32) ushort_t lA[2][16][32][16];
    __shared__ __align__(32) ushort_t lB[2][8][32][16];

    v8f acc[8] = {};
    const size_t bmB = (size_t)b * KK * NN;
    const size_t xB = (size_t)b * NN * CC;

    auto stageA = [&](int nchunk, int buf) {
#pragma unroll
        for (int i = 0; i < 2; i++) {
            int idx = t + i * 512;
            int kk = idx >> 2, cg = idx & 3;
            const ushort_t* src = bmm + bmB + (size_t)kk * NN + nchunk + cg * 8;
            ushort_t* dst = &lA[buf][kk >> 4][(kk & 15) + ((cg & 1) << 4)][(cg >> 1) << 3];
            ASYNC_CP16(src, dst);
        }
    };
    auto stageB = [&](int nchunk, int buf) {
        int nl = t >> 4;               // K index 0..31
        int cg = t & 15;               // 16 groups of 8 c
        int c0 = cg * 8;
        uint4 v = *(const uint4*)(xbf + xB + (size_t)(nchunk + nl) * CC + cbase + c0);
        const ushort_t* pv = (const ushort_t*)&v;
        int half = (nl >> 3) & 1, slot = (nl & 7) + ((nl >> 4) << 3);
#pragma unroll
        for (int j = 0; j < 8; j++) {
            int cl = c0 + j;
            lB[buf][cl >> 4][(half << 4) + (cl & 15)][slot] = pv[j];
        }
    };

    stageA(n0, 0); stageB(n0, 0);
    ASYNC_FENCE();
    __syncthreads();

    for (int nn = 0; nn < NN / 16; nn += 32) {
        int buf = (nn >> 5) & 1;
        int nchunk = n0 + nn;
        if (nn + 32 < NN / 16) {
            stageA(nchunk + 32, buf ^ 1);
            stageB(nchunk + 32, buf ^ 1);
            __builtin_prefetch(bmm + bmB + (size_t)(t >> 1) * NN + nchunk + 64, 0, 3);
        }
        v16bf a = *(const v16bf*)&lA[buf][wave][lane][0];
#pragma unroll
        for (int cs = 0; cs < 8; cs++) {
            v16bf bb = *(const v16bf*)&lB[buf][cs][lane][0];
            acc[cs] = __builtin_amdgcn_wmma_f32_16x16x32_bf16(
                false, a, false, bb, (short)0, acc[cs], false, false);
        }
        ASYNC_FENCE();
        __syncthreads();
    }

    int kbase = wave * 16 + (lane >> 4) * 8;
    int ccol = lane & 15;
#pragma unroll
    for (int cs = 0; cs < 8; cs++) {
        int c = cbase + cs * 16 + ccol;
#pragma unroll
        for (int r = 0; r < 8; r++)
            atomicAdd(&protoAcc[((size_t)b * KK + kbase + r) * CC + c], acc[cs][r]);
    }
}

// normalize proto, emit f32 + bf16 copies, and next stage's entropy term
__global__ void k_finalize(const float* __restrict__ protoAcc, const ushort_t* __restrict__ bmm,
                           float* __restrict__ protoF, ushort_t* __restrict__ protoH,
                           float* __restrict__ ent) {
    int b = blockIdx.y, kk = blockIdx.x, t = threadIdx.x;
    const ushort_t* row = bmm + ((size_t)b * KK + kk) * NN;
    float s = 0.f;
    for (int i = t; i < NN; i += 256) s += bf2f(row[i]);
    __shared__ float sh[256];
    sh[t] = s; __syncthreads();
    for (int st = 128; st > 0; st >>= 1) { if (t < st) sh[t] += sh[t + st]; __syncthreads(); }
    float denom = sh[0] + 1e-6f;
    __syncthreads();
    float es = 0.f;
    for (int c = t; c < CC; c += 256) {
        size_t pi = ((size_t)b * KK + kk) * CC + c;
        float p = protoAcc[pi] / denom;
        protoF[pi] = p; protoH[pi] = f2bf(p);
        es += p * __logf(p);
    }
    sh[t] = es; __syncthreads();
    for (int st = 128; st > 0; st >>= 1) { if (t < st) sh[t] += sh[t + st]; __syncthreads(); }
    if (t == 0) ent[b * KK + kk] = sh[0];
}

__global__ void k_copy_f32(const float* __restrict__ src, float* __restrict__ dst, size_t n) {
    size_t i = (size_t)blockIdx.x * blockDim.x + threadIdx.x;
    if (i < n) dst[i] = src[i];
}

// ------------------------------------------------------------------- launcher
extern "C" void kernel_launch(void* const* d_in, const int* in_sizes, int n_in,
                              void* d_out, int out_size, void* d_ws, size_t ws_size,
                              hipStream_t stream) {
    const float* x = (const float*)d_in[0];
    char* w = (char*)d_ws;

    ushort_t* lxbf   = (ushort_t*)(w + 0);            // 128 MB  log(x) bf16 (b,n,c)
    ushort_t* xbf    = (ushort_t*)(w + 134217728);    // 128 MB  x bf16 (b,n,c)
    ushort_t* bmm    = (ushort_t*)(w + 268435456);    //  64 MB  b_map bf16 (b,k,n)
    float*    protoF = (float*)   (w + 335544320);    //   4 MB
    ushort_t* protoH = (ushort_t*)(w + 339738624);    //   2 MB
    float*    protoA = (float*)   (w + 341835776);    //   4 MB  accumulation
    float*    ent    = (float*)   (w + 346030080);    //   8 KB
    float*    pmin   = (float*)   (w + 346038272);
    float*    pmax   = pmin + BB * 256;
    float*    mnv    = pmax + BB * 256;
    float*    invv   = mnv + BB;

    int*   cl   = (int*)d_out;                        // cluster (b,n) as int32 bits
    float* outP = (float*)d_out + (size_t)BB * NN;    // proto (b,k,c) f32

    k_minmax_part<<<dim3(256, BB), 256, 0, stream>>>(x, pmin, pmax);
    k_minmax_fin<<<BB, 256, 0, stream>>>(pmin, pmax, mnv, invv);
    k_transform<<<dim3((CC * NN) / 256, BB), 256, 0, stream>>>(x, mnv, invv, xbf, lxbf, protoF, protoH);
    k_zero_u16<<<(int)(((size_t)BB * KK * NN + 255) / 256), 256, 0, stream>>>(bmm, (size_t)BB * KK * NN);
    k_ent<<<dim3(KK, BB), 256, 0, stream>>>(protoF, ent);

    for (int s = 0; s < STAGES; s++) {
        k_dist<<<dim3(NN / 128, BB), 512, 0, stream>>>(protoH, lxbf, ent, cl, bmm);
        k_zero_f32<<<(int)(((size_t)BB * KK * CC + 255) / 256), 256, 0, stream>>>(protoA, (size_t)BB * KK * CC);
        k_update<<<dim3(CC / 128, 16, BB), 512, 0, stream>>>(bmm, xbf, protoA);
        k_finalize<<<dim3(KK, BB), 256, 0, stream>>>(protoA, bmm, protoF, protoH, ent);
    }
    k_copy_f32<<<(int)(((size_t)BB * KK * CC + 255) / 256), 256, 0, stream>>>(protoF, outP, (size_t)BB * KK * CC);
}
